// MoE_32341103739481
// MI455X (gfx1250) — matrix operand order repaired and checked by reading, not compiled, not run on hardware
//
#include <hip/hip_runtime.h>

#define N_TOK 8192
#define DIM   1024
#define NEXP  8
#define HID   256

#define TM 64            // token tile per block
#define XS 40            // xbuf row stride (ushort elems), padded vs 32
#define WSR 40           // weight buffer row stride (K elems per col), padded vs 32
#define HSR 264          // h buffer row stride, padded vs 256
#define TILES_MAX (N_TOK / TM)   // 128

typedef __attribute__((ext_vector_type(16))) _Float16 v16h;
typedef __attribute__((ext_vector_type(8)))  float    v8f;

union FragU  { v16h v; uint4 q[2]; };
union Pack8  { unsigned short s[8];  uint4 q;    };

#if defined(__has_builtin)
#if __has_builtin(__builtin_amdgcn_global_load_async_to_lds_b128) && \
    __has_builtin(__builtin_amdgcn_s_wait_asynccnt)
#define USE_ASYNC_LDS 1
#endif
#endif
#ifndef USE_ASYNC_LDS
#define USE_ASYNC_LDS 0
#endif

#if USE_ASYNC_LDS
typedef int v4i_ __attribute__((vector_size(16)));
typedef __attribute__((address_space(1))) v4i_* g4p;   // global int4-vector ptr
typedef __attribute__((address_space(3))) v4i_* l4p;   // LDS int4-vector ptr
#define WAIT_ASYNC() __builtin_amdgcn_s_wait_asynccnt(0)
#else
#define WAIT_ASYNC()
#endif

__device__ __forceinline__ unsigned short f2h_bits(float f) {
  _Float16 h = (_Float16)f;                  // RNE f32->f16
  return __builtin_bit_cast(unsigned short, h);
}

// A fragment 16x32 f16 (ISA 7.12.2): lane L holds row M=L%16;
// elems 0..7 -> K = klo+0..7, elems 8..15 -> K = klo+16..23, klo = (L<16)?0:8.
__device__ __forceinline__ v16h load_a_frag(const unsigned short* base, int stride,
                                            int lane, int kbase) {
  int row = lane & 15;
  int klo = (lane < 16) ? 0 : 8;
  const unsigned short* p = base + row * stride + kbase + klo;
  FragU f;
  f.q[0] = *(const uint4*)(p);        // K klo..klo+7
  f.q[1] = *(const uint4*)(p + 16);   // K klo+16..klo+23
  return f.v;
}

// B fragment 32x16 f16: lane L holds col N=L%16; elems 0..7 -> K = koff+0..7,
// elems 8..15 -> K = koff+16..23, koff = (L<16)?0:8.  Weights live K-contiguous in LDS: wb[col*WSR + k].
__device__ __forceinline__ v16h load_b_frag(const unsigned short* wb, int lane, int col0) {
  int col = col0 + (lane & 15);
  int koff = (lane < 16) ? 0 : 8;
  const unsigned short* p = wb + col * WSR + koff;
  FragU f;
  f.q[0] = *(const uint4*)(p);
  f.q[1] = *(const uint4*)(p + 16);
  return f.v;
}

// Stage a 256-col x 32-K chunk of a pre-transposed f16 weight matrix
// (layout [Nn][Kfull], K contiguous) into wb[col*WSR + k]. One col per thread:
// a straight 64B copy -> async global->LDS when available.
__device__ __forceinline__ void stage_w(unsigned short* wb, const unsigned short* wtsrc,
                                        int Kfull, int k0, int nbase, int tid) {
  const unsigned short* wp = wtsrc + (size_t)(nbase + tid) * Kfull + k0;
  unsigned short* o = wb + tid * WSR;
#if USE_ASYNC_LDS
  __builtin_amdgcn_global_load_async_to_lds_b128((g4p)(wp),      (l4p)(o),      0, 0);
  __builtin_amdgcn_global_load_async_to_lds_b128((g4p)(wp + 8),  (l4p)(o + 8),  0, 0);
  __builtin_amdgcn_global_load_async_to_lds_b128((g4p)(wp + 16), (l4p)(o + 16), 0, 0);
  __builtin_amdgcn_global_load_async_to_lds_b128((g4p)(wp + 24), (l4p)(o + 24), 0, 0);
#else
  uint4 q0 = *(const uint4*)(wp);
  uint4 q1 = *(const uint4*)(wp + 8);
  uint4 q2 = *(const uint4*)(wp + 16);
  uint4 q3 = *(const uint4*)(wp + 24);
  *(uint4*)(o)      = q0;
  *(uint4*)(o + 8)  = q1;
  *(uint4*)(o + 16) = q2;
  *(uint4*)(o + 24) = q3;
#endif
}

// Gather one 8-float segment of x, convert f32->f16, store 16B to LDS.
__device__ __forceinline__ void stage_x_chunk(unsigned short* xb, const float* xrow,
                                              int k0, int sr, int sseg) {
  float4 a4 = *(const float4*)(xrow + k0);
  float4 c4 = *(const float4*)(xrow + k0 + 4);
  Pack8 u;
  u.s[0] = f2h_bits(a4.x); u.s[1] = f2h_bits(a4.y);
  u.s[2] = f2h_bits(a4.z); u.s[3] = f2h_bits(a4.w);
  u.s[4] = f2h_bits(c4.x); u.s[5] = f2h_bits(c4.y);
  u.s[6] = f2h_bits(c4.z); u.s[7] = f2h_bits(c4.w);
  *(uint4*)&xb[sr * XS + sseg] = u.q;
}

// ---------------------------------------------------------------------------
__global__ void zero_counts_kernel(int* counts) {
  if (threadIdx.x < NEXP) counts[threadIdx.x] = 0;
}

// Tiled transpose + cast: src [E][K][Nn] f32 -> dst [E][Nn][K] f16.
__global__ __launch_bounds__(256) void transpose_cast_kernel(
    const float* __restrict__ src, unsigned short* __restrict__ dst, int K, int Nn) {
  __shared__ float tile[32][33];
  int ntk = K >> 5, ntn = Nn >> 5;
  int bid = blockIdx.x;
  int e   = bid / (ntk * ntn);
  int rem = bid % (ntk * ntn);
  int tk  = rem / ntn, tn = rem % ntn;
  int tx = threadIdx.x & 31, ty = threadIdx.x >> 5;   // 32 x 8
  const float* s = src + ((size_t)e * K + tk * 32) * Nn + tn * 32;
#pragma unroll
  for (int i = 0; i < 4; i++)
    tile[ty + i * 8][tx] = s[(size_t)(ty + i * 8) * Nn + tx];
  __syncthreads();
  unsigned short* d = dst + ((size_t)e * Nn + tn * 32) * K + tk * 32;
#pragma unroll
  for (int i = 0; i < 4; i++)
    d[(size_t)(ty + i * 8) * K + tx] = f2h_bits(tile[tx][ty + i * 8]);
}

// Gating: one wave (32 lanes) per token.
__global__ __launch_bounds__(256) void gate_kernel(
    const float* __restrict__ x, const float* __restrict__ Wg,
    const float* __restrict__ bg, int* __restrict__ counts,
    int* __restrict__ list, float* __restrict__ gates) {
  int lane = threadIdx.x & 31;
  int wid  = threadIdx.x >> 5;
  int n = blockIdx.x * 8 + wid;
  if (n >= N_TOK) return;

  float p[NEXP];
#pragma unroll
  for (int e = 0; e < NEXP; e++) p[e] = 0.f;
  const float* xr = x + (size_t)n * DIM;
  for (int d = lane; d < DIM; d += 32) {
    float xv = xr[d];
    const float* wr = Wg + (size_t)d * NEXP;
#pragma unroll
    for (int e = 0; e < NEXP; e++) p[e] += xv * wr[e];
  }
#pragma unroll
  for (int e = 0; e < NEXP; e++) {
#pragma unroll
    for (int off = 16; off > 0; off >>= 1) p[e] += __shfl_xor(p[e], off, 32);
  }
  if (lane == 0) {
    float m = -1e30f;
#pragma unroll
    for (int e = 0; e < NEXP; e++) { p[e] += bg[e]; m = fmaxf(m, p[e]); }
    float s = 0.f;
#pragma unroll
    for (int e = 0; e < NEXP; e++) { p[e] = __expf(p[e] - m); s += p[e]; }
    float inv = 1.f / s;
#pragma unroll
    for (int e = 0; e < NEXP; e++) p[e] *= inv;
    int i1 = 0; float v1 = p[0];
#pragma unroll
    for (int e = 1; e < NEXP; e++) if (p[e] > v1) { v1 = p[e]; i1 = e; }
    int i2 = -1; float v2 = -1.f;
#pragma unroll
    for (int e = 0; e < NEXP; e++) if (e != i1 && p[e] > v2) { v2 = p[e]; i2 = e; }
    int p1 = atomicAdd(&counts[i1], 1);
    list [i1 * N_TOK + p1] = n;                 // slot 0 (top-1)
    gates[i1 * N_TOK + p1] = v1;
    int p2 = atomicAdd(&counts[i2], 1);
    list [i2 * N_TOK + p2] = n | (1 << 16);     // slot 1 (top-2)
    gates[i2 * N_TOK + p2] = v2;
  }
}

// Fused 3-layer expert MLP over one tile of gathered tokens. Double-buffered
// K-chunk staging: stage chunk kc+1 while computing chunk kc (1 barrier/iter).
__global__ __launch_bounds__(256) void moe_expert_kernel(
    const float* __restrict__ x,
    const float* __restrict__ b1, const float* __restrict__ b2,
    const float* __restrict__ b3,
    const unsigned short* __restrict__ W1t, const unsigned short* __restrict__ W2t,
    const unsigned short* __restrict__ W3t,
    const int* __restrict__ counts, const int* __restrict__ list,
    const float* __restrict__ gates, float* __restrict__ slots) {
  __shared__ __attribute__((aligned(16))) unsigned short xbuf[2 * TM * XS];
  __shared__ __attribute__((aligned(16))) unsigned short wbuf[2 * HID * WSR];
  __shared__ __attribute__((aligned(16))) unsigned short hbuf[TM * HSR];
  __shared__ int   tokbuf[TM];
  __shared__ float gbuf[TM];

  int e    = blockIdx.x & (NEXP - 1);
  int tile = blockIdx.x >> 3;
  int cnt  = counts[e];
  int n0   = tile * TM;
  if (n0 >= cnt) return;

  int tid   = threadIdx.x;
  int lane  = tid & 31;
  int wid   = tid >> 5;
  int mrow0 = (wid & 3) * 16;       // 4 M-strips of 16 rows
  int nb    = (wid >> 2) * 128;     // 2 N-halves of 128 cols

  if (tid < TM) {
    int idx = n0 + tid;
    if (idx < cnt) { tokbuf[tid] = list[e * N_TOK + idx]; gbuf[tid] = gates[e * N_TOK + idx]; }
    else           { tokbuf[tid] = 0;                      gbuf[tid] = 0.f; }
  }
  __syncthreads();

  const unsigned short* W1te = W1t + (size_t)e * HID * DIM;   // [HID][DIM]
  const unsigned short* W2te = W2t + (size_t)e * HID * HID;   // [HID][HID]
  const unsigned short* W3te = W3t + (size_t)e * DIM * HID;   // [DIM][HID]

  int sr   = tid >> 2;
  int sseg = (tid & 3) * 8;
  int stok = tokbuf[sr] & 0xFFFF;
  const float* xrow = x + (size_t)stok * DIM + sseg;

  const v8f zero8 = {0.f, 0.f, 0.f, 0.f, 0.f, 0.f, 0.f, 0.f};
  v8f acc[8];

  // ---------------- Layer 1: h1 = relu(x @ W1[e] + b1[e]) ----------------
#pragma unroll
  for (int t = 0; t < 8; t++) acc[t] = zero8;
  stage_x_chunk(xbuf, xrow, 0, sr, sseg);
  stage_w(wbuf, W1te, DIM, 0, 0, tid);
  for (int kc = 0; kc < DIM / 32; kc++) {
    unsigned short* xb = xbuf + (kc & 1) * (TM * XS);
    unsigned short* wb = wbuf + (kc & 1) * (HID * WSR);
    WAIT_ASYNC();
    __syncthreads();
    if (kc + 1 < DIM / 32) {
      stage_x_chunk(xbuf + ((kc + 1) & 1) * (TM * XS), xrow, (kc + 1) * 32, sr, sseg);
      stage_w(wbuf + ((kc + 1) & 1) * (HID * WSR), W1te, DIM, (kc + 1) * 32, 0, tid);
    }
    v16h a = load_a_frag(xb + mrow0 * XS, XS, lane, 0);
#pragma unroll
    for (int t = 0; t < 8; t++) {
      v16h b = load_b_frag(wb, lane, nb + t * 16);
      acc[t] = __builtin_amdgcn_wmma_f32_16x16x32_f16(false, a, false, b,
                                                      (short)0, acc[t], false, false);
    }
  }
  {
    int rb = mrow0 + ((lane < 16) ? 0 : 8);
#pragma unroll
    for (int t = 0; t < 8; t++) {
      int col = nb + t * 16 + (lane & 15);
      float bias = b1[e * HID + col];
#pragma unroll
      for (int r = 0; r < 8; r++) {
        float v = fmaxf(acc[t][r] + bias, 0.f);
        hbuf[(rb + r) * HSR + col] = f2h_bits(v);
      }
    }
  }

  // ---------------- Layer 2: h2 = relu(h1 @ W2[e] + b2[e]) ----------------
#pragma unroll
  for (int t = 0; t < 8; t++) acc[t] = zero8;
  stage_w(wbuf, W2te, HID, 0, 0, tid);          // wbuf[0] free (last L1 read was buf 1)
  for (int kc = 0; kc < HID / 32; kc++) {
    unsigned short* wb = wbuf + (kc & 1) * (HID * WSR);
    WAIT_ASYNC();
    __syncthreads();                             // kc==0: also publishes hbuf (h1)
    if (kc + 1 < HID / 32)
      stage_w(wbuf + ((kc + 1) & 1) * (HID * WSR), W2te, HID, (kc + 1) * 32, 0, tid);
    v16h a = load_a_frag(&hbuf[mrow0 * HSR], HSR, lane, kc * 32);
#pragma unroll
    for (int t = 0; t < 8; t++) {
      v16h b = load_b_frag(wb, lane, nb + t * 16);
      acc[t] = __builtin_amdgcn_wmma_f32_16x16x32_f16(false, a, false, b,
                                                      (short)0, acc[t], false, false);
    }
  }
  __syncthreads();                               // all h1 reads done -> safe to overwrite
  {
    int rb = mrow0 + ((lane < 16) ? 0 : 8);
#pragma unroll
    for (int t = 0; t < 8; t++) {
      int col = nb + t * 16 + (lane & 15);
      float bias = b2[e * HID + col];
#pragma unroll
      for (int r = 0; r < 8; r++) {
        float v = fmaxf(acc[t][r] + bias, 0.f);
        hbuf[(rb + r) * HSR + col] = f2h_bits(v);   // reuse hbuf for h2
      }
    }
  }

  // ------- Layer 3: out_slot = gate * relu(h2 @ W3[e] + b3[e]) -------
  for (int c = 0; c < 4; c++) {
    int cbase = c * 256;
#pragma unroll
    for (int t = 0; t < 8; t++) acc[t] = zero8;
    stage_w(wbuf, W3te, HID, 0, cbase, tid);
    for (int kc = 0; kc < HID / 32; kc++) {
      unsigned short* wb = wbuf + (kc & 1) * (HID * WSR);
      WAIT_ASYNC();
      __syncthreads();                           // c==0,kc==0: also publishes h2
      if (kc + 1 < HID / 32)
        stage_w(wbuf + ((kc + 1) & 1) * (HID * WSR), W3te, HID, (kc + 1) * 32, cbase, tid);
      v16h a = load_a_frag(&hbuf[mrow0 * HSR], HSR, lane, kc * 32);
#pragma unroll
      for (int t = 0; t < 8; t++) {
        v16h b = load_b_frag(wb, lane, nb + t * 16);
        acc[t] = __builtin_amdgcn_wmma_f32_16x16x32_f16(false, a, false, b,
                                                        (short)0, acc[t], false, false);
      }
    }
    int rb = mrow0 + ((lane < 16) ? 0 : 8);
#pragma unroll
    for (int t = 0; t < 8; t++) {
      int col = cbase + nb + t * 16 + (lane & 15);
      float bias = b3[e * DIM + col];
#pragma unroll
      for (int r = 0; r < 8; r++) {
        int row = rb + r;
        if (n0 + row < cnt) {
          float v = fmaxf(acc[t][r] + bias, 0.f) * gbuf[row];
          int entry = tokbuf[row];
          int tok  = entry & 0xFFFF;
          int slot = entry >> 16;
          slots[((size_t)slot * N_TOK + tok) * DIM + col] = v;
        }
      }
    }
  }
}

// out = slot0 + slot1 (every token has exactly 2 assignments)
__global__ void combine_kernel(const float* __restrict__ slots, float* __restrict__ out) {
  int i = blockIdx.x * blockDim.x + threadIdx.x;   // float4 index, grid is exact
  const float4* a = (const float4*)slots;
  const float4* b = (const float4*)(slots + (size_t)N_TOK * DIM);
  float4 va = a[i], vb = b[i];
  float4 o; o.x = va.x + vb.x; o.y = va.y + vb.y; o.z = va.z + vb.z; o.w = va.w + vb.w;
  ((float4*)out)[i] = o;
}

// ---------------------------------------------------------------------------
extern "C" void kernel_launch(void* const* d_in, const int* in_sizes, int n_in,
                              void* d_out, int out_size, void* d_ws, size_t ws_size,
                              hipStream_t stream) {
  const float* x  = (const float*)d_in[0];
  const float* Wg = (const float*)d_in[1];
  const float* bg = (const float*)d_in[2];
  const float* W1 = (const float*)d_in[3];
  const float* b1 = (const float*)d_in[4];
  const float* W2 = (const float*)d_in[5];
  const float* b2 = (const float*)d_in[6];
  const float* W3 = (const float*)d_in[7];
  const float* b3 = (const float*)d_in[8];
  float* out = (float*)d_out;

  char* ws = (char*)d_ws;
  size_t off = 0;
  auto alloc = [&](size_t bytes) {
    char* p = ws + off;
    off += (bytes + 255) & ~(size_t)255;
    return p;
  };
  const int W1n = NEXP * DIM * HID;
  const int W2n = NEXP * HID * HID;
  const int W3n = NEXP * HID * DIM;
  unsigned short* W1t = (unsigned short*)alloc((size_t)W1n * 2);  // [E][HID][DIM]
  unsigned short* W2t = (unsigned short*)alloc((size_t)W2n * 2);  // [E][HID][HID]
  unsigned short* W3t = (unsigned short*)alloc((size_t)W3n * 2);  // [E][DIM][HID]
  int*   counts = (int*)  alloc(NEXP * sizeof(int));
  int*   list   = (int*)  alloc((size_t)NEXP * N_TOK * sizeof(int));
  float* gates  = (float*)alloc((size_t)NEXP * N_TOK * sizeof(float));
  float* slots  = (float*)alloc((size_t)2 * N_TOK * DIM * sizeof(float));

  zero_counts_kernel<<<1, 32, 0, stream>>>(counts);
  // W1 [E][DIM][HID] -> W1t [E][HID][DIM]
  transpose_cast_kernel<<<NEXP * (DIM / 32) * (HID / 32), 256, 0, stream>>>(W1, W1t, DIM, HID);
  // W2 [E][HID][HID] -> W2t [E][HID][HID] (transposed)
  transpose_cast_kernel<<<NEXP * (HID / 32) * (HID / 32), 256, 0, stream>>>(W2, W2t, HID, HID);
  // W3 [E][HID][DIM] -> W3t [E][DIM][HID]
  transpose_cast_kernel<<<NEXP * (HID / 32) * (DIM / 32), 256, 0, stream>>>(W3, W3t, HID, DIM);
  gate_kernel<<<N_TOK / 8, 256, 0, stream>>>(x, Wg, bg, counts, list, gates);
  moe_expert_kernel<<<NEXP * TILES_MAX, 256, 0, stream>>>(
      x, b1, b2, b3, W1t, W2t, W3t, counts, list, gates, slots);
  combine_kernel<<<(N_TOK * DIM / 4) / 256, 256, 0, stream>>>(slots, out);
}
